// LSTMAutoEncoder_6356551598372
// MI455X (gfx1250) — matrix operand order repaired and checked by reading, not compiled, not run on hardware
//
#include <hip/hip_runtime.h>
#include <hip/hip_bf16.h>

// ---------------------------------------------------------------------------
// LSTM auto-encoder, MI455X (gfx1250, wave32, WMMA).
//   B=4096, L=360, H=128.
//   enc step: h0=0  ->  gates = x0@enc_W_ih.T + b   (no W_hh term)
//   dec:      xproj = x0@dec_W_ih.T + b  is constant across the 360 steps;
//             only h@dec_W_hh.T (193 GFLOP) is sequential.
//   Recurrence (round 3): the recurrence is LDS-bandwidth bound (every 32-row
//   group streams all 128 KB of W_hh from LDS each step), so spread it over
//   as many WGPs as possible: 128 blocks of 2 waves, each wave owning half
//   the hidden units (disjoint halves of W_hh -> no duplicated B traffic).
//   152 KB LDS/block allows two blocks per WGP. M=32 rows per wave-tile so
//   each 1KB B-fragment feeds two WMMAs; c state in LDS fragment-major;
//   xproj pre-swizzled to C-fragment layout (coalesced b128 init from L2).
// ---------------------------------------------------------------------------

typedef __attribute__((ext_vector_type(16))) _Float16 v16h;
typedef __attribute__((ext_vector_type(8)))  float    v8f;

#define LENGTH 360
#define HIDDEN 128
#define BATCH  4096
#define KPAD   384      // 360 padded to multiple of 32
#define NGATES 512      // 4*HIDDEN
#define NPAD_FC 368     // 360 padded to multiple of 16
#define DEC_ROWS 32     // batch rows per decoder block

union AFragU { v16h v; uint4 q[2]; };
union CFragU { v8f v; uint4 q[2]; };

// A fragment: 16x32 f16, rows M=lane%16.
// lanes 0-15 : v0..v3 = K 0..7,  v4..v7 = K 16..23
// lanes 16-31: v0..v3 = K 8..15, v4..v7 = K 24..31
__device__ __forceinline__ v16h load_a_frag(const _Float16* A, int ld, int kt, int lane) {
    const int m  = lane & 15;
    const int kb = kt * 32 + ((lane < 16) ? 0 : 8);
    const _Float16* p = A + m * ld + kb;
    AFragU f;
    f.q[0] = *reinterpret_cast<const uint4*>(p);
    f.q[1] = *reinterpret_cast<const uint4*>(p + 16);
    return f.v;
}

// B fragment: 32x16 f16 (K x N), W stored row-major as (N x K).
// lane holds column N=lane%16; lanes 0-15 K +0..7,+16..23, lanes 16-31 K +8..15,+24..31.
__device__ __forceinline__ v16h load_b_frag(const _Float16* W, int ld, int n0, int kt, int lane) {
    const int n  = n0 + (lane & 15);
    const int kb = kt * 32 + ((lane < 16) ? 0 : 8);
    const _Float16* p = W + n * ld + kb;
    AFragU f;
    f.q[0] = *reinterpret_cast<const uint4*>(p);
    f.q[1] = *reinterpret_cast<const uint4*>(p + 16);
    return f.v;
}

__device__ __forceinline__ float sigmoidf_(float x) {
    return 1.0f / (1.0f + __expf(-x));
}

// --------------------------- packing kernels -------------------------------

__global__ void cvt_pad_f16_kernel(const float* __restrict__ src, _Float16* __restrict__ dst,
                                   int srows, int scols, int drows, int dcols) {
    int i = blockIdx.x * blockDim.x + threadIdx.x;
    if (i >= drows * dcols) return;
    int r = i / dcols, c = i % dcols;
    float v = (r < srows && c < scols) ? src[r * scols + c] : 0.0f;
    dst[i] = (_Float16)v;
}

__global__ void bias_sum_kernel(const float* __restrict__ a, const float* __restrict__ b,
                                float* __restrict__ out, int n) {
    int i = blockIdx.x * blockDim.x + threadIdx.x;
    if (i < n) out[i] = a[i] + b[i];
}

// Swizzle xproj (B x 512, f32) into C-fragment-major layout:
//   xf[((g*64 + f)*32 + lane)*8 + r],  g = 32-row group, f = (ht*4 + gate)*2 + mt
//   element = xproj[(g*32 + mt*16 + (lane<16?0:8) + r) * 512 + gate*128 + ht*16 + (lane&15)]
__global__ void xproj_swizzle_kernel(const float* __restrict__ xp, float* __restrict__ xf) {
    int i = blockIdx.x * blockDim.x + threadIdx.x;
    if (i >= BATCH * NGATES) return;
    int r = i & 7;
    int l = (i >> 3) & 31;
    int f = (i >> 8) & 63;
    int g = i >> 14;
    int mt = f & 1, gate = (f >> 1) & 3, ht = f >> 3;
    int row = g * 32 + mt * 16 + ((l < 16) ? 0 : 8) + r;
    int col = gate * HIDDEN + ht * 16 + (l & 15);
    xf[i] = xp[(size_t)row * NGATES + col];
}

// ------------------------- generic WMMA tile GEMM --------------------------
// C[M x nvalid] = A[M x Kp](f16) @ W[Np x Kp](f16)^T + bias  (f32 out)
__global__ __launch_bounds__(256) void gemm_bias_wmma(
        const _Float16* __restrict__ A, const _Float16* __restrict__ W,
        const float* __restrict__ bias, float* __restrict__ C,
        int M, int Np, int Kp, int ldc, int nvalid) {
    const int lane = threadIdx.x & 31;
    const int wid  = threadIdx.x >> 5;
    const int numNT = Np >> 4;
    const int numMT = M >> 4;
    const int tile = blockIdx.x * 8 + wid;
    if (tile >= numMT * numNT) return;
    const int mt = tile / numNT;
    const int nt = tile % numNT;

    const int ncol = nt * 16 + (lane & 15);
    const int mbase = mt * 16 + ((lane < 16) ? 0 : 8);

    v8f acc;
    const float bv = (ncol < nvalid) ? bias[ncol] : 0.0f;
#pragma unroll
    for (int r = 0; r < 8; ++r) acc[r] = bv;

    const _Float16* Atile = A + (size_t)mt * 16 * Kp;
    const int nkt = Kp >> 5;
    for (int kt = 0; kt < nkt; ++kt) {
        v16h a = load_a_frag(Atile, Kp, kt, lane);
        v16h b = load_b_frag(W, Kp, nt * 16, kt, lane);
        acc = __builtin_amdgcn_wmma_f32_16x16x32_f16(false, a, false, b, (short)0, acc, false, false);
    }

    if (ncol < nvalid) {
#pragma unroll
        for (int r = 0; r < 8; ++r)
            C[(size_t)(mbase + r) * ldc + ncol] = acc[r];
    }
}

// ------------------------ encoder activation (h0=0) ------------------------
__global__ void enc_act_kernel(const float* __restrict__ xproj,
                               _Float16* __restrict__ h16, float* __restrict__ c32) {
    int i = blockIdx.x * blockDim.x + threadIdx.x;
    if (i >= BATCH * HIDDEN) return;
    int b = i / HIDDEN, j = i % HIDDEN;
    const float* g = xproj + (size_t)b * NGATES;
    float gi = g[j], gg = g[j + 2 * HIDDEN], go = g[j + 3 * HIDDEN];
    float c = sigmoidf_(gi) * tanhf(gg);        // f gate multiplies c0 == 0
    float h = sigmoidf_(go) * tanhf(c);
    c32[i] = c;
    h16[i] = (_Float16)h;
}

// --------------------------- decoder recurrence ----------------------------
// block: 64 threads = 2 waves, one 32-row batch group per block (128 blocks).
// wave wid owns hidden tiles ht = wid*4 .. wid*4+3 (disjoint half of W_hh).
// LDS: W_hh f16 128 KB + h f16 8 KB + c f32 16 KB = 152 KB (2 blocks/WGP).
__global__ __launch_bounds__(64) void lstm_decoder_kernel(
        const float* __restrict__ xfrag,   // fragment-major xproj, (128,64,32,8) f32
        const _Float16* __restrict__ whh,  // (512, 128) f16
        const float* __restrict__ c_in,    // (B, 128) f32
        _Float16* __restrict__ h16,        // (B, 128) f16, in/out
        int steps) {
    extern __shared__ _Float16 smem[];
    _Float16* sW = smem;                                  // 512*128 halves
    _Float16* sH = smem + NGATES * HIDDEN;                // 32*128 halves
    float*    sC = (float*)(smem + NGATES * HIDDEN + DEC_ROWS * HIDDEN);  // 16*32*8 f32

    const int tid  = threadIdx.x;
    const int lane = tid & 31;
    const int wid  = tid >> 5;          // 0 or 1
    const int r0   = blockIdx.x * DEC_ROWS;

    // cooperative fill: W_hh (8192 x 16B) and this block's 32 h rows (512 x 16B)
    {
        const uint4* gw = reinterpret_cast<const uint4*>(whh);
        uint4* lw = reinterpret_cast<uint4*>(sW);
        for (int i = tid; i < (NGATES * HIDDEN) / 8; i += 64) lw[i] = gw[i];
        const uint4* gh = reinterpret_cast<const uint4*>(h16 + (size_t)r0 * HIDDEN);
        uint4* lh = reinterpret_cast<uint4*>(sH);
        for (int i = tid; i < (DEC_ROWS * HIDDEN) / 8; i += 64) lh[i] = gh[i];
    }

    const int ml   = (lane < 16) ? 0 : 8;
    const int ncol = lane & 15;
    const int ht0  = wid * 4;           // this wave's first hidden tile

    // initial c into LDS fragment-major: sC[(ht*2+mt)*32 + lane][8]
#pragma unroll
    for (int htl = 0; htl < 4; ++htl) {
        const int ht = ht0 + htl;
#pragma unroll
        for (int mt = 0; mt < 2; ++mt)
#pragma unroll
            for (int r = 0; r < 8; ++r)
                sC[((ht * 2 + mt) * 32 + lane) * 8 + r] =
                    c_in[(size_t)(r0 + mt * 16 + ml + r) * HIDDEN + ht * 16 + ncol];
    }
    __syncthreads();

    const float* xf = xfrag + (size_t)blockIdx.x * 64 * 32 * 8;

    for (int t = 0; t < steps; ++t) {
        // A fragments of h (2 M-tiles x 4 K-tiles): full 128-wide h, both waves
        v16h a[2][4];
#pragma unroll
        for (int mt = 0; mt < 2; ++mt)
#pragma unroll
            for (int kt = 0; kt < 4; ++kt)
                a[mt][kt] = load_a_frag(sH + mt * 16 * HIDDEN, HIDDEN, kt, lane);

#pragma unroll 1
        for (int htl = 0; htl < 4; ++htl) {
            const int ht = ht0 + htl;
            const int nI = ht * 16;
            v8f acc[4][2];   // [gate][mtile]
            // accumulator init: coalesced b128 pairs from fragment-major xproj (L2)
#pragma unroll
            for (int gate = 0; gate < 4; ++gate)
#pragma unroll
                for (int mt = 0; mt < 2; ++mt) {
                    CFragU u;
                    const uint4* p = reinterpret_cast<const uint4*>(
                        xf + (size_t)(((ht * 4 + gate) * 2 + mt) * 32 + lane) * 8);
                    u.q[0] = p[0];
                    u.q[1] = p[1];
                    acc[gate][mt] = u.v;
                }
            // each B fragment feeds two WMMAs (M reuse); 8 independent chains
#pragma unroll
            for (int kt = 0; kt < 4; ++kt) {
#pragma unroll
                for (int gate = 0; gate < 4; ++gate) {
                    v16h b = load_b_frag(sW, HIDDEN, gate * HIDDEN + nI, kt, lane);
                    acc[gate][0] = __builtin_amdgcn_wmma_f32_16x16x32_f16(
                        false, a[0][kt], false, b, (short)0, acc[gate][0], false, false);
                    acc[gate][1] = __builtin_amdgcn_wmma_f32_16x16x32_f16(
                        false, a[1][kt], false, b, (short)0, acc[gate][1], false, false);
                }
            }
            // gate math in f32; c in LDS (per-lane private), h back as f16
#pragma unroll
            for (int mt = 0; mt < 2; ++mt) {
                float* cp = sC + ((ht * 2 + mt) * 32 + lane) * 8;
#pragma unroll
                for (int r = 0; r < 8; ++r) {
                    float ii = sigmoidf_(acc[0][mt][r]);
                    float ff = sigmoidf_(acc[1][mt][r]);
                    float gv = tanhf(acc[2][mt][r]);
                    float oo = sigmoidf_(acc[3][mt][r]);
                    float cn = ff * cp[r] + ii * gv;
                    cp[r] = cn;
                    sH[(mt * 16 + ml + r) * HIDDEN + nI + ncol] = (_Float16)(oo * tanhf(cn));
                }
            }
        }
        __syncthreads();   // cross-wave: full h must be visible for next A loads
    }

    // write final h back to global for the FC GEMM
    {
        const uint4* lh = reinterpret_cast<const uint4*>(sH);
        uint4* gh = reinterpret_cast<uint4*>(h16 + (size_t)r0 * HIDDEN);
        for (int i = tid; i < (DEC_ROWS * HIDDEN) / 8; i += 64) gh[i] = lh[i];
    }
}

// ------------------------------- launcher ----------------------------------

extern "C" void kernel_launch(void* const* d_in, const int* in_sizes, int n_in,
                              void* d_out, int out_size, void* d_ws, size_t ws_size,
                              hipStream_t stream) {
    const float* x        = (const float*)d_in[0];   // (4096,1,360)
    const float* enc_W_ih = (const float*)d_in[1];   // (512,360)
    // d_in[2] = enc_W_hh: unused (h0 == 0)
    const float* enc_b_ih = (const float*)d_in[3];
    const float* enc_b_hh = (const float*)d_in[4];
    const float* dec_W_ih = (const float*)d_in[5];   // (512,360)
    const float* dec_W_hh = (const float*)d_in[6];   // (512,128)
    const float* dec_b_ih = (const float*)d_in[7];
    const float* dec_b_hh = (const float*)d_in[8];
    const float* fc_W     = (const float*)d_in[9];   // (360,128)
    const float* fc_b     = (const float*)d_in[10];  // (360,)
    float* out = (float*)d_out;                      // (4096,1,360)

    char* ws = (char*)d_ws;
    size_t off = 0;
    auto alloc = [&](size_t bytes) { char* p = ws + off; off = (off + bytes + 255) & ~(size_t)255; return p; };

    _Float16* x16    = (_Float16*)alloc((size_t)BATCH * KPAD * 2);
    _Float16* encW16 = (_Float16*)alloc((size_t)NGATES * KPAD * 2);
    _Float16* decW16 = (_Float16*)alloc((size_t)NGATES * KPAD * 2);
    _Float16* whh16  = (_Float16*)alloc((size_t)NGATES * HIDDEN * 2);
    _Float16* fcW16  = (_Float16*)alloc((size_t)NPAD_FC * HIDDEN * 2);
    float*    encB   = (float*)alloc(NGATES * 4);
    float*    decB   = (float*)alloc(NGATES * 4);
    float*    xpEnc  = (float*)alloc((size_t)BATCH * NGATES * 4);
    float*    xpDec  = (float*)alloc((size_t)BATCH * NGATES * 4);
    float*    xfDec  = (float*)alloc((size_t)BATCH * NGATES * 4);   // fragment-major
    float*    c32    = (float*)alloc((size_t)BATCH * HIDDEN * 4);
    _Float16* h16    = (_Float16*)alloc((size_t)BATCH * HIDDEN * 2);
    (void)ws_size;

    const int TB = 256;
    auto grid1d = [&](long n) { return dim3((unsigned)((n + TB - 1) / TB)); };

    // 1) convert + pad to f16
    cvt_pad_f16_kernel<<<grid1d((long)BATCH * KPAD), TB, 0, stream>>>(x, x16, BATCH, LENGTH, BATCH, KPAD);
    cvt_pad_f16_kernel<<<grid1d((long)NGATES * KPAD), TB, 0, stream>>>(enc_W_ih, encW16, NGATES, LENGTH, NGATES, KPAD);
    cvt_pad_f16_kernel<<<grid1d((long)NGATES * KPAD), TB, 0, stream>>>(dec_W_ih, decW16, NGATES, LENGTH, NGATES, KPAD);
    cvt_pad_f16_kernel<<<grid1d((long)NGATES * HIDDEN), TB, 0, stream>>>(dec_W_hh, whh16, NGATES, HIDDEN, NGATES, HIDDEN);
    cvt_pad_f16_kernel<<<grid1d((long)NPAD_FC * HIDDEN), TB, 0, stream>>>(fc_W, fcW16, LENGTH, HIDDEN, NPAD_FC, HIDDEN);
    bias_sum_kernel<<<grid1d(NGATES), TB, 0, stream>>>(enc_b_ih, enc_b_hh, encB, NGATES);
    bias_sum_kernel<<<grid1d(NGATES), TB, 0, stream>>>(dec_b_ih, dec_b_hh, decB, NGATES);

    // 2) x-projections: (4096 x 384) @ (512 x 384)^T + bias
    {
        int tiles = (BATCH / 16) * (NGATES / 16);
        int blocks = (tiles + 7) / 8;
        gemm_bias_wmma<<<blocks, 256, 0, stream>>>(x16, encW16, encB, xpEnc, BATCH, NGATES, KPAD, NGATES, NGATES);
        gemm_bias_wmma<<<blocks, 256, 0, stream>>>(x16, decW16, decB, xpDec, BATCH, NGATES, KPAD, NGATES, NGATES);
    }

    // 3) encoder activations -> h (f16), c (f32); swizzle dec xproj to fragments
    enc_act_kernel<<<grid1d((long)BATCH * HIDDEN), TB, 0, stream>>>(xpEnc, h16, c32);
    xproj_swizzle_kernel<<<grid1d((long)BATCH * NGATES), TB, 0, stream>>>(xpDec, xfDec);

    // 4) 360-step decoder recurrence: 128 blocks x 2 waves, W_hh + h + c in LDS
    {
        size_t lds = (size_t)NGATES * HIDDEN * 2      // W_hh f16  128 KB
                   + (size_t)DEC_ROWS * HIDDEN * 2    // h f16       8 KB
                   + (size_t)16 * 32 * 8 * 4;         // c f32      16 KB
        lstm_decoder_kernel<<<BATCH / DEC_ROWS, 64, lds, stream>>>(
            xfDec, whh16, c32, h16, LENGTH);
    }

    // 5) pred = h @ fc_W^T + fc_b  -> (4096, 360) f32 (N padded to 368, guarded)
    {
        int tiles = (BATCH / 16) * (NPAD_FC / 16);
        int blocks = (tiles + 7) / 8;
        gemm_bias_wmma<<<blocks, 256, 0, stream>>>(h16, fcW16, fc_b, out, BATCH, NPAD_FC, HIDDEN, LENGTH, LENGTH);
    }
    (void)in_sizes; (void)n_in; (void)out_size;
}
